// MultiHeadAttentionLayer_26594437497101
// MI455X (gfx1250) — matrix operand order repaired and checked by reading, not compiled, not run on hardware
//
#include <hip/hip_runtime.h>
#include <hip/hip_bf16.h>

// MI455X (gfx1250) multi-head attention.
// Roofline: ~86 GFLOP vs ~100 MB HBM traffic -> matrix-compute bound; use f16 WMMA
// with f32 accumulation. Flash kernel shares K/V chunks across all 8 waves of a
// workgroup via LDS (async global->LDS copies, ASYNCcnt) for 8x L2-traffic reuse.

#define D_MODEL   512
#define NUM_HEADS 8
#define BATCH     2
#define SEQ       4096
#define DK        64
#define KCHUNK    64   // keys staged per iteration (8 KB K + 8 KB V in LDS)

typedef __attribute__((ext_vector_type(16))) _Float16 v16h;
typedef __attribute__((ext_vector_type(8)))  _Float16 v8h;
typedef __attribute__((ext_vector_type(8)))  float    v8f;
typedef int b128i __attribute__((vector_size(16)));   // matches builtin's V4i param

#ifndef __has_builtin
#define __has_builtin(x) 0
#endif

#if __has_builtin(__builtin_amdgcn_global_load_async_to_lds_b128)
#define HAVE_ASYNC_LDS 1
#else
#define HAVE_ASYNC_LDS 0
#endif

#define AS1 __attribute__((address_space(1)))
#define AS3 __attribute__((address_space(3)))

#if HAVE_ASYNC_LDS
// Signature (from compiler diagnostic): (v4i* global_src, v4i* lds_dst, Ii off, Ii cpol).
// Same INST_OFFSET is added to BOTH the global and LDS addresses (ISA 08, §4.4).
// Integer round-trip casts sidestep addrspacecast restrictions; for LDS the low
// 32 bits of the flat address are the LDS byte offset (ISA 00, aperture rules).
#define ASYNC_CP16(g, l, off)                                                 \
    __builtin_amdgcn_global_load_async_to_lds_b128(                           \
        (AS1 b128i*)(unsigned long long)(const void*)(g),                     \
        (AS3 b128i*)(unsigned int)(unsigned long long)(const void*)(l),       \
        (off), 0)
#endif

#if __has_builtin(__builtin_amdgcn_s_wait_asynccnt)
#define WAIT_ASYNC() __builtin_amdgcn_s_wait_asynccnt(0)
#else
#define WAIT_ASYNC() asm volatile("s_wait_asynccnt 0x0" ::: "memory")
#endif

// D = A(16x32 f16) * B(32x16 f16) + C(16x16 f32)
__device__ __forceinline__ v8f wmma16(v16h a, v16h b, v8f c) {
    return __builtin_amdgcn_wmma_f32_16x16x32_f16(
        false, a, false, b, (short)0, c, false, false);
}

// ---- Fragment loaders (wave32; contiguous k order of this code, not the ISA 7.12.2 table order) ----
// A 16x32 (MxK): row = row0+(lane&15); K = k0 + h + (lane>>4)*16
// => one contiguous 16-half run per lane.
__device__ __forceinline__ v16h load_afrag_h(const _Float16* base, int ld,
                                             int row0, int k0, int lrow, int lhalf) {
    const _Float16* p = base + (size_t)(row0 + lrow) * ld + k0 + lhalf * 16;
    v8h lo = *(const v8h*)p;
    v8h hi = *(const v8h*)(p + 8);
    v16h a;
#pragma unroll
    for (int i = 0; i < 8; ++i) { a[i] = lo[i]; a[i + 8] = hi[i]; }
    return a;
}

__device__ __forceinline__ v16h load_afrag_f(const float* __restrict__ base, int ld,
                                             int row0, int k0, int lrow, int lhalf) {
    const float* p = base + (size_t)(row0 + lrow) * ld + k0 + lhalf * 16;
    v16h a;
#pragma unroll
    for (int i = 0; i < 8; ++i) {
        a[i]     = (_Float16)p[i];
        a[i + 8] = (_Float16)p[8 + i];
    }
    return a;
}

// B 32x16 (KxN) from a row-major [N,K] matrix (NT gemm):
// N = col0+(lane&15); K = k0 + h + (lane>>4)*16 => one contiguous 32B run per lane.
__device__ __forceinline__ v16h load_bfrag_h(const _Float16* base, int ld,
                                             int col0, int k0, int lrow, int lhalf) {
    const _Float16* p = base + (size_t)(col0 + lrow) * ld + k0 + lhalf * 16;
    return *(const v16h*)p;
}

__device__ __forceinline__ v16h load_bfrag_f(const float* __restrict__ base, int ld,
                                             int col0, int k0, int lrow, int lhalf) {
    const float* p = base + (size_t)(col0 + lrow) * ld + k0 + lhalf * 16;
    v16h b;
#pragma unroll
    for (int i = 0; i < 16; ++i) b[i] = (_Float16)p[i];
    return b;
}

// ---------------- Kernel 1: fused Q/K/V projection (NT GEMM, fp32->f16) -------------
// grid = (512, 3); 8 waves/block; each wave computes a 16x64 strip (4 C tiles) so one
// A fragment feeds 4 WMMAs (4x less activation traffic than 1 tile/wave).
__global__ __launch_bounds__(256) void proj_qkv_kernel(
    const float* __restrict__ q_in, const float* __restrict__ k_in, const float* __restrict__ v_in,
    const float* __restrict__ Wq, const float* __restrict__ Wk, const float* __restrict__ Wv,
    const float* __restrict__ bq, const float* __restrict__ bk, const float* __restrict__ bv,
    _Float16* __restrict__ Qh, _Float16* __restrict__ Kh, _Float16* __restrict__ Vt)
{
    const int lane = threadIdx.x & 31, wave = threadIdx.x >> 5;
    const int lrow = lane & 15, lhalf = lane >> 4;
    const int which = blockIdx.y;

    const float* X    = which == 0 ? q_in : which == 1 ? k_in : v_in;
    const float* W    = which == 0 ? Wq   : which == 1 ? Wk   : Wv;
    const float* bias = which == 0 ? bq   : which == 1 ? bk   : bv;

    const int t    = blockIdx.x * 8 + wave;   // 4096 strips: 512 m-tiles x 8 n-groups
    const int row0 = (t >> 3) * 16;
    const int col0 = (t & 7) * 64;

    v8f acc0 = {}, acc1 = {}, acc2 = {}, acc3 = {};
#pragma unroll 2
    for (int k0 = 0; k0 < D_MODEL; k0 += 32) {
        const v16h a = load_afrag_f(X, D_MODEL, row0, k0, lrow, lhalf);
        v16h b = load_bfrag_f(W, D_MODEL, col0, k0, lrow, lhalf);
        acc0 = wmma16(a, b, acc0);
        b = load_bfrag_f(W, D_MODEL, col0 + 16, k0, lrow, lhalf);
        acc1 = wmma16(a, b, acc1);
        b = load_bfrag_f(W, D_MODEL, col0 + 32, k0, lrow, lhalf);
        acc2 = wmma16(a, b, acc2);
        b = load_bfrag_f(W, D_MODEL, col0 + 48, k0, lrow, lhalf);
        acc3 = wmma16(a, b, acc3);
    }

#pragma unroll
    for (int f = 0; f < 4; ++f) {
        const v8f* accp = f == 0 ? &acc0 : f == 1 ? &acc1 : f == 2 ? &acc2 : &acc3;
        const int n = col0 + f * 16 + lrow;
        const float bn = bias[n];
        const int head = n >> 6, dim = n & 63;
#pragma unroll
        for (int r = 0; r < 8; ++r) {
            const int m = row0 + r + 8 * lhalf;       // C-layout: row = r + 8*(lane>>4)
            const int b = m >> 12, s = m & (SEQ - 1);
            const float val = (*accp)[r] + bn;
            if (which < 2) {
                _Float16* Out = which == 0 ? Qh : Kh;
                Out[(((size_t)(b * NUM_HEADS + head)) * SEQ + s) * DK + dim] = (_Float16)val;
            } else {
                Vt[(((size_t)(b * NUM_HEADS + head)) * DK + dim) * SEQ + s] = (_Float16)val;
            }
        }
    }
}

// ---------------- Kernel 2: flash attention with workgroup K/V sharing --------------
// Block = 8 waves = 128 q-rows of one (b,h). K/V chunks staged to LDS once per block
// (async global->LDS when available), consumed by all 8 waves -> 8x L2-reuse.
// Reference semantics: attn = softmax(QK^T) / sqrt(d_k) (NO logit pre-scale) =>
// fold 0.125 into the final 1/rowsum normalization.
__global__ __launch_bounds__(256) void flash_attn_kernel(
    const _Float16* __restrict__ Qh, const _Float16* __restrict__ Kh,
    const _Float16* __restrict__ Vt, _Float16* __restrict__ Xh)
{
    __shared__ _Float16 Ks[KCHUNK * DK];        // [key][dim]  8 KB
    __shared__ _Float16 Vs[DK * KCHUNK];        // [dim][key]  8 KB
    __shared__ _Float16 Plds[8][16 * KCHUNK];   // per-wave P  16 KB

    const int tid  = threadIdx.x;
    const int lane = tid & 31, wave = tid >> 5;
    const int lrow = lane & 15, lhalf = lane >> 4;

    const int bh = blockIdx.x >> 5;                       // 32 blocks per (b,h)
    const int q0 = ((blockIdx.x & 31) * 8 + wave) * 16;   // this wave's 16 q-rows

    const _Float16* Qp = Qh + (size_t)bh * SEQ * DK;
    const _Float16* Kp = Kh + (size_t)bh * SEQ * DK;
    const _Float16* Vp = Vt + (size_t)bh * DK * SEQ;
    _Float16* P = Plds[wave];

    // Q fragments stay in registers for the whole key loop.
    const v16h qa0 = load_afrag_h(Qp, DK, q0, 0,  lrow, lhalf);
    const v16h qa1 = load_afrag_h(Qp, DK, q0, 32, lrow, lhalf);

    v8f acc0 = {}, acc1 = {}, acc2 = {}, acc3 = {};
    float rm[8], rl[8];
#pragma unroll
    for (int r = 0; r < 8; ++r) { rm[r] = -1e30f; rl[r] = 0.0f; }

    const int vrow = tid >> 3, vseg = tid & 7;   // V staging: 8 x 16B segs per 128B row

    for (int kc = 0; kc < SEQ; kc += KCHUNK) {
        __syncthreads();                         // previous chunk fully consumed
        // ---- stage K (8 KB contiguous) and V (64 rows x 128 B) into LDS ----
        const _Float16* gK  = Kp + (size_t)kc * DK;
        const _Float16* gV0 = Vp + (size_t)vrow * SEQ + kc;
        const _Float16* gV1 = Vp + (size_t)(vrow + 32) * SEQ + kc;
#if HAVE_ASYNC_LDS
        {
            const char* gKc = (const char*)gK + tid * 16;
            char*       lKc = (char*)Ks + tid * 16;
            ASYNC_CP16(gKc, lKc, 0);
            ASYNC_CP16(gKc, lKc, 4096);
            ASYNC_CP16((const char*)gV0 + vseg * 16,
                       (char*)Vs + vrow * 128 + vseg * 16, 0);
            ASYNC_CP16((const char*)gV1 + vseg * 16,
                       (char*)Vs + (vrow + 32) * 128 + vseg * 16, 0);
        }
        WAIT_ASYNC();
#else
        {
            v8h t0 = *(const v8h*)((const char*)gK + tid * 16);
            v8h t1 = *(const v8h*)((const char*)gK + tid * 16 + 4096);
            v8h t2 = *(const v8h*)((const char*)gV0 + vseg * 16);
            v8h t3 = *(const v8h*)((const char*)gV1 + vseg * 16);
            *(v8h*)((char*)Ks + tid * 16)            = t0;
            *(v8h*)((char*)Ks + tid * 16 + 4096)     = t1;
            *(v8h*)((char*)Vs + vrow * 128 + vseg * 16)        = t2;
            *(v8h*)((char*)Vs + (vrow + 32) * 128 + vseg * 16) = t3;
        }
#endif
        __syncthreads();                         // staged K/V visible to all waves

        // ---- scores: S(16x64) = Q(16x64) @ K_chunk^T, four 16x16 C tiles ----
        v8f s0 = {}, s1 = {}, s2 = {}, s3 = {};
        {
            v16h kb;
            kb = load_bfrag_h(Ks, DK, 0,  0,  lrow, lhalf); s0 = wmma16(qa0, kb, s0);
            kb = load_bfrag_h(Ks, DK, 0,  32, lrow, lhalf); s0 = wmma16(qa1, kb, s0);
            kb = load_bfrag_h(Ks, DK, 16, 0,  lrow, lhalf); s1 = wmma16(qa0, kb, s1);
            kb = load_bfrag_h(Ks, DK, 16, 32, lrow, lhalf); s1 = wmma16(qa1, kb, s1);
            kb = load_bfrag_h(Ks, DK, 32, 0,  lrow, lhalf); s2 = wmma16(qa0, kb, s2);
            kb = load_bfrag_h(Ks, DK, 32, 32, lrow, lhalf); s2 = wmma16(qa1, kb, s2);
            kb = load_bfrag_h(Ks, DK, 48, 0,  lrow, lhalf); s3 = wmma16(qa0, kb, s3);
            kb = load_bfrag_h(Ks, DK, 48, 32, lrow, lhalf); s3 = wmma16(qa1, kb, s3);
        }
        // ---- online softmax; each C row lives in 16 lanes of one half ----
#pragma unroll
        for (int r = 0; r < 8; ++r) {
            float mx = fmaxf(fmaxf(s0[r], s1[r]), fmaxf(s2[r], s3[r]));
            mx = fmaxf(mx, __shfl_xor(mx, 1));
            mx = fmaxf(mx, __shfl_xor(mx, 2));
            mx = fmaxf(mx, __shfl_xor(mx, 4));
            mx = fmaxf(mx, __shfl_xor(mx, 8));
            const float nm   = fmaxf(rm[r], mx);
            const float corr = __expf(rm[r] - nm);
            const float p0 = __expf(s0[r] - nm);
            const float p1 = __expf(s1[r] - nm);
            const float p2 = __expf(s2[r] - nm);
            const float p3 = __expf(s3[r] - nm);
            s0[r] = p0; s1[r] = p1; s2[r] = p2; s3[r] = p3;
            float rs = (p0 + p1) + (p2 + p3);
            rs += __shfl_xor(rs, 1);
            rs += __shfl_xor(rs, 2);
            rs += __shfl_xor(rs, 4);
            rs += __shfl_xor(rs, 8);
            rl[r] = rl[r] * corr + rs;
            rm[r] = nm;
            acc0[r] *= corr; acc1[r] *= corr; acc2[r] *= corr; acc3[r] *= corr;
        }
        // ---- P: C-layout f32 -> f16 A-fragments via this wave's LDS region ----
#pragma unroll
        for (int r = 0; r < 8; ++r) {
            const int row = r + 8 * lhalf;
            P[row * KCHUNK +      lrow] = (_Float16)s0[r];
            P[row * KCHUNK + 16 + lrow] = (_Float16)s1[r];
            P[row * KCHUNK + 32 + lrow] = (_Float16)s2[r];
            P[row * KCHUNK + 48 + lrow] = (_Float16)s3[r];
        }
        asm volatile("" ::: "memory");   // per-wave DS ops are in-order in HW
        const v16h pa0 = load_afrag_h(P, KCHUNK, 0, 0,  lrow, lhalf);
        const v16h pa1 = load_afrag_h(P, KCHUNK, 0, 32, lrow, lhalf);
        asm volatile("" ::: "memory");
        // ---- O(16x64) += P(16x64) @ V_chunk(64x64) ----
        {
            v16h vb;
            vb = load_bfrag_h(Vs, KCHUNK, 0,  0,  lrow, lhalf); acc0 = wmma16(pa0, vb, acc0);
            vb = load_bfrag_h(Vs, KCHUNK, 0,  32, lrow, lhalf); acc0 = wmma16(pa1, vb, acc0);
            vb = load_bfrag_h(Vs, KCHUNK, 16, 0,  lrow, lhalf); acc1 = wmma16(pa0, vb, acc1);
            vb = load_bfrag_h(Vs, KCHUNK, 16, 32, lrow, lhalf); acc1 = wmma16(pa1, vb, acc1);
            vb = load_bfrag_h(Vs, KCHUNK, 32, 0,  lrow, lhalf); acc2 = wmma16(pa0, vb, acc2);
            vb = load_bfrag_h(Vs, KCHUNK, 32, 32, lrow, lhalf); acc2 = wmma16(pa1, vb, acc2);
            vb = load_bfrag_h(Vs, KCHUNK, 48, 0,  lrow, lhalf); acc3 = wmma16(pa0, vb, acc3);
            vb = load_bfrag_h(Vs, KCHUNK, 48, 32, lrow, lhalf); acc3 = wmma16(pa1, vb, acc3);
        }
    }

    // ---- normalize (fold in /sqrt(d_k) = *0.125) and store f16 X ----
    const int b = bh >> 3, head = bh & 7;
#pragma unroll
    for (int r = 0; r < 8; ++r) {
        const float inv  = 0.125f / rl[r];
        const int   srow = q0 + r + 8 * lhalf;
        const size_t rowbase = ((size_t)b * SEQ + srow) * D_MODEL + head * DK;
        Xh[rowbase + 0  + lrow] = (_Float16)(acc0[r] * inv);
        Xh[rowbase + 16 + lrow] = (_Float16)(acc1[r] * inv);
        Xh[rowbase + 32 + lrow] = (_Float16)(acc2[r] * inv);
        Xh[rowbase + 48 + lrow] = (_Float16)(acc3[r] * inv);
    }
}

// ---------------- Kernel 3: output projection (f16 X @ Wo^T + bo -> fp32) -----------
__global__ __launch_bounds__(256) void out_proj_kernel(
    const _Float16* __restrict__ Xh, const float* __restrict__ Wo,
    const float* __restrict__ bo, float* __restrict__ Out)
{
    const int lane = threadIdx.x & 31, wave = threadIdx.x >> 5;
    const int lrow = lane & 15, lhalf = lane >> 4;
    const int t    = blockIdx.x * 8 + wave;
    const int row0 = (t >> 3) * 16;
    const int col0 = (t & 7) * 64;

    v8f acc0 = {}, acc1 = {}, acc2 = {}, acc3 = {};
#pragma unroll 2
    for (int k0 = 0; k0 < D_MODEL; k0 += 32) {
        const v16h a = load_afrag_h(Xh, D_MODEL, row0, k0, lrow, lhalf);
        v16h b = load_bfrag_f(Wo, D_MODEL, col0, k0, lrow, lhalf);
        acc0 = wmma16(a, b, acc0);
        b = load_bfrag_f(Wo, D_MODEL, col0 + 16, k0, lrow, lhalf);
        acc1 = wmma16(a, b, acc1);
        b = load_bfrag_f(Wo, D_MODEL, col0 + 32, k0, lrow, lhalf);
        acc2 = wmma16(a, b, acc2);
        b = load_bfrag_f(Wo, D_MODEL, col0 + 48, k0, lrow, lhalf);
        acc3 = wmma16(a, b, acc3);
    }
#pragma unroll
    for (int f = 0; f < 4; ++f) {
        const v8f* accp = f == 0 ? &acc0 : f == 1 ? &acc1 : f == 2 ? &acc2 : &acc3;
        const int n = col0 + f * 16 + lrow;
        const float bn = bo[n];
#pragma unroll
        for (int r = 0; r < 8; ++r) {
            const int m = row0 + r + 8 * lhalf;
            Out[(size_t)m * D_MODEL + n] = (*accp)[r] + bn;
        }
    }
}

extern "C" void kernel_launch(void* const* d_in, const int* in_sizes, int n_in,
                              void* d_out, int out_size, void* d_ws, size_t ws_size,
                              hipStream_t stream) {
    const float* query = (const float*)d_in[0];
    const float* key_  = (const float*)d_in[1];
    const float* value = (const float*)d_in[2];
    const float* Wq    = (const float*)d_in[3];
    const float* bq    = (const float*)d_in[4];
    const float* Wk    = (const float*)d_in[5];
    const float* bk    = (const float*)d_in[6];
    const float* Wv    = (const float*)d_in[7];
    const float* bv    = (const float*)d_in[8];
    const float* Wo    = (const float*)d_in[9];
    const float* bo    = (const float*)d_in[10];
    float* out = (float*)d_out;

    // Workspace: 4 f16 buffers of B*S*D = 4M halves (8 MB) each = 32 MB total.
    const size_t elems = (size_t)BATCH * SEQ * D_MODEL;
    _Float16* Qh = (_Float16*)d_ws;
    _Float16* Kh = Qh + elems;
    _Float16* Vt = Kh + elems;
    _Float16* Xh = Vt + elems;

    dim3 blk(256);
    proj_qkv_kernel<<<dim3(512, 3), blk, 0, stream>>>(query, key_, value,
                                                      Wq, Wk, Wv, bq, bk, bv,
                                                      Qh, Kh, Vt);
    flash_attn_kernel<<<dim3(512), blk, 0, stream>>>(Qh, Kh, Vt, Xh);
    out_proj_kernel<<<dim3(512), blk, 0, stream>>>(Xh, Wo, bo, out);
}
